// RoIHead_48430051230374
// MI455X (gfx1250) — matrix operand order repaired and checked by reading, not compiled, not run on hardware
//
#include <hip/hip_runtime.h>
#include <hip/hip_bf16.h>

// ---------------------------------------------------------------------------
// RoI head for MI455X (gfx1250, wave32, WMMA + async global->LDS).
// All GEMMs run on v_wmma_f32_16x16x32_bf16 (bf16 A/B, f32 accum).
// ---------------------------------------------------------------------------

#define NPTS   4096
#define CFEAT  128
#define NROI   128
#define GRID   6
#define KNN    16
#define COUT   64
#define MTOT   (NROI * GRID * GRID * GRID)   // 27648
#define FLATD  (GRID * GRID * GRID * COUT)   // 13824
#define R2     0.64f

// Async global->LDS path (CDNA5): guarded so the build can't regress.
#if defined(__has_builtin)
#if __has_builtin(__builtin_amdgcn_global_load_async_to_lds_b128) && \
    __has_builtin(__builtin_amdgcn_s_wait_asynccnt)
#define USE_ASYNC_LDS 1
#endif
#endif
#ifndef USE_ASYNC_LDS
#define USE_ASYNC_LDS 0
#endif

typedef __bf16 v16bf __attribute__((ext_vector_type(16)));
typedef float  v8f   __attribute__((ext_vector_type(8)));
typedef int    v4i   __attribute__((ext_vector_type(4)));

__device__ __forceinline__ unsigned short f2bf(float f) {
  unsigned u = __float_as_uint(f);
  unsigned r = (u + 0x7FFFu + ((u >> 16) & 1u)) >> 16;   // RNE
  return (unsigned short)r;
}
__device__ __forceinline__ float bf2f(unsigned short h) {
  return __uint_as_float(((unsigned)h) << 16);
}

// ---------------------------------------------------------------------------
// feats f32 -> bf16 table (done once; 1 MB, L2-resident; feeds async DMA).
// ---------------------------------------------------------------------------
__global__ void k_feats2bf(const float* __restrict__ feats, unsigned short* __restrict__ out) {
  int i = blockIdx.x * 256 + threadIdx.x;
  if (i < NPTS * CFEAT) out[i] = f2bf(feats[i]);
}

// ---------------------------------------------------------------------------
// Pack an f32 [Kdim x Ndim] row-major weight into bf16 WMMA B-fragments.
// Tile t = kt*ntiles + nt holds 512 bf16: lane l (l/16 = K-half, l%16 = col),
// element r (0..15): k = kt*32 + (l/16)*16 + r, n = nt*16 + l%16 (contiguous k order, same as load_afrag).
// ---------------------------------------------------------------------------
__global__ void k_frag(const float* __restrict__ W, unsigned short* __restrict__ out,
                       int Kdim, int Ndim, int ntiles) {
  int t  = blockIdx.x;
  int kt = t / ntiles, nt = t % ntiles;
  for (int e = threadIdx.x; e < 512; e += blockDim.x) {
    int lane = e >> 4, r = e & 15;
    int k = kt * 32 + (lane >> 4) * 16 + r;
    int n = nt * 16 + (lane & 15);
    float v = (k < Kdim) ? W[(size_t)k * Ndim + n] : 0.0f;
    out[(size_t)t * 512 + e] = f2bf(v);
  }
}

// w_m1 fragments with the A-column permutation [feats(128) | rel(3) | pad]:
// A col k<128 -> w_m1 row k+3 ; A col 128..130 -> w_m1 row k-128 ; else 0.
__global__ void k_frag_m1(const float* __restrict__ W, unsigned short* __restrict__ out) {
  int t  = blockIdx.x;                    // 5 ktiles x 4 ntiles
  int kt = t / 4, nt = t % 4;
  for (int e = threadIdx.x; e < 512; e += blockDim.x) {
    int lane = e >> 4, r = e & 15;
    int k = kt * 32 + (lane >> 4) * 16 + r;
    int n = nt * 16 + (lane & 15);
    int src = (k < 128) ? (k + 3) : ((k < 131) ? (k - 128) : -1);
    float v = (src >= 0) ? W[(size_t)src * COUT + n] : 0.0f;
    out[(size_t)t * 512 + e] = f2bf(v);
  }
}

// ---------------------------------------------------------------------------
// Grid points: get_dense_grid_points + rotate_z + center shift.
// ---------------------------------------------------------------------------
__global__ void k_gridpoints(const float* __restrict__ rois, float* __restrict__ gp) {
  int m = blockIdx.x * 256 + threadIdx.x;
  if (m >= MTOT) return;
  int r = m / 216, t = m % 216;
  int gi = t / 36, gj = (t / 6) % 6, gk = t % 6;
  const float* ro = rois + r * 7;
  float sx = ro[3], sy = ro[4], sz = ro[5];
  float lx = (gi + 0.5f) * (1.0f / GRID) * sx - 0.5f * sx;
  float ly = (gj + 0.5f) * (1.0f / GRID) * sy - 0.5f * sy;
  float lz = (gk + 0.5f) * (1.0f / GRID) * sz - 0.5f * sz;
  float c = __cosf(ro[6]), s = __sinf(ro[6]);
  gp[m * 3 + 0] = lx * c - ly * s + ro[0];
  gp[m * 3 + 1] = lx * s + ly * c + ro[1];
  gp[m * 3 + 2] = lz + ro[2];
}

// ---------------------------------------------------------------------------
// Neighbor search: one wave32 per grid point, ballot+popcount compaction.
// ---------------------------------------------------------------------------
__global__ void k_neighbors(const float* __restrict__ gp, const float* __restrict__ xyz,
                            int* __restrict__ idxA, int* __restrict__ emp) {
  int wave = threadIdx.x >> 5, lane = threadIdx.x & 31;
  int m = blockIdx.x * 8 + wave;
  float gx = gp[m * 3], gy = gp[m * 3 + 1], gz = gp[m * 3 + 2];
  int* dst = idxA + (size_t)m * KNN;
  int cnt = 0;
  int first = 0;
  bool haveFirst = false;
  for (int c = 0; c < NPTS; c += 32) {
    int p = c + lane;
    float dx = xyz[p * 3] - gx, dy = xyz[p * 3 + 1] - gy, dz = xyz[p * 3 + 2] - gz;
    bool pred = (dx * dx + dy * dy + dz * dz) < R2;
    unsigned mask = (unsigned)__ballot(pred);
    int pos = cnt + __popc(mask & ((1u << lane) - 1u));
    if (pred && pos < KNN) dst[pos] = p;
    if (!haveFirst && mask) { first = c + __ffs(mask) - 1; haveFirst = true; }
    cnt += __popc(mask);
    if (cnt >= KNN) break;     // uniform: cnt derived from ballot
  }
  if (lane < KNN && lane >= cnt) dst[lane] = haveFirst ? first : 0;
  if (lane == 0) emp[m] = haveFirst ? 0 : 1;
}

// ---------------------------------------------------------------------------
// A-fragment assembly: 16-bit A 16x32, contiguous k order (same as k_frag B packing).
// Lane l: row = l%16, khalf = l/16; dword j: k = khalf*16 + 2*j.
// ---------------------------------------------------------------------------
__device__ __forceinline__ v16bf load_afrag(const unsigned short* src, int stride,
                                            int row, int khalf, int kbase) {
  union { v16bf v; unsigned u[8]; } a;
#pragma unroll
  for (int j = 0; j < 8; ++j) {
    int k = kbase + khalf * 16 + (j << 1);
    a.u[j] = *(const unsigned*)(src + (size_t)row * stride + k);
  }
  return a.v;
}

// ---------------------------------------------------------------------------
// Per-grid-point MLP + maxpool. One wave per grid point (8 waves/block).
// A-tile column layout: [feats(128) | rel(3) | pad(29)] so each neighbor's
// bf16 feature row is a 16B-aligned 256B block -> async DMA'd straight to LDS.
// ---------------------------------------------------------------------------
__global__ void k_pointmlp(const float* __restrict__ gp, const float* __restrict__ xyz,
                           const unsigned short* __restrict__ pfeat,  // [NPTS][128] bf16
                           const int* __restrict__ idxA, const int* __restrict__ emp,
                           const unsigned short* __restrict__ fragM1,   // 5x4 tiles
                           const unsigned short* __restrict__ fragM2,   // 2x4 tiles
                           unsigned short* __restrict__ pooled) {
  __shared__ unsigned short sg[8][16 * 160];   // A tile, K padded -> 160
  __shared__ unsigned short sh[8][16 * 64];    // hidden tile
  __shared__ int sidx[8][KNN];

  int wave = threadIdx.x >> 5, lane = threadIdx.x & 31;
  int m = blockIdx.x * 8 + wave;

  if (lane < KNN) sidx[wave][lane] = idxA[(size_t)m * KNN + lane];
  float gx = gp[m * 3], gy = gp[m * 3 + 1], gz = gp[m * 3 + 2];
  int em = emp[m];
  __syncthreads();

#if USE_ASYNC_LDS
  {
    // Each lane DMAs one 128B half-row of bf16 feats into the LDS A tile:
    // row = lane/2, seg = lane&1 -> cols seg*64..seg*64+63. 8x b128 per lane.
    typedef __attribute__((address_space(1))) v4i GV4;   // global int4
    typedef __attribute__((address_space(3))) v4i LV4;   // LDS int4
    int r = lane >> 1, seg = lane & 1;
    int p = sidx[wave][r];
    const unsigned short* gsrc = pfeat + (size_t)p * CFEAT + seg * 64;
    unsigned short* ldst = &sg[wave][r * 160 + seg * 64];
#pragma unroll
    for (int i = 0; i < 8; ++i) {
      __builtin_amdgcn_global_load_async_to_lds_b128(
          (GV4*)(unsigned long long)(uintptr_t)(gsrc + i * 8),
          (LV4*)(unsigned)(uintptr_t)(ldst + i * 8), 0, 0);
    }
    // Overlap: rel-xyz + zero pad (cols 128..159) while the DMA runs.
    for (int e = lane; e < 16 * 32; e += 32) {
      int rr = e >> 5, c = 128 + (e & 31);
      int pp = sidx[wave][rr];
      float val = 0.0f;
      if (c < 131) val = xyz[pp * 3 + (c - 128)] - (c == 128 ? gx : (c == 129 ? gy : gz));
      sg[wave][rr * 160 + c] = f2bf(val);
    }
    __builtin_amdgcn_s_wait_asynccnt(0);
  }
#else
  // Sync fallback: build the whole A tile through VGPRs.
  for (int e = lane; e < 16 * 160; e += 32) {
    int r = e / 160, c = e % 160;
    int p = sidx[wave][r];
    float val = 0.0f;
    if (c < 128)      val = bf2f(pfeat[(size_t)p * CFEAT + c]);
    else if (c < 131) val = xyz[p * 3 + (c - 128)] - (c == 128 ? gx : (c == 129 ? gy : gz));
    sg[wave][e] = f2bf(val);
  }
#endif
  __syncthreads();

  int row = lane & 15, khalf = lane >> 4;

  // Layer 1: 4 n-tiles x 5 k-tiles.
#pragma unroll
  for (int nt = 0; nt < 4; ++nt) {
    v8f acc = {};
#pragma unroll
    for (int kt = 0; kt < 5; ++kt) {
      v16bf a = load_afrag(sg[wave], 160, row, khalf, kt * 32);
      v16bf b = *(const v16bf*)(fragM1 + (((size_t)kt * 4 + nt) << 9) + lane * 16);
      acc = __builtin_amdgcn_wmma_f32_16x16x32_bf16(false, a, false, b,
                                                    (short)0, acc, false, false);
    }
    int n = nt * 16 + row;
#pragma unroll
    for (int v = 0; v < 8; ++v)
      sh[wave][(v + 8 * khalf) * 64 + n] = f2bf(fmaxf(acc[v], 0.0f));
  }
  __syncthreads();

  // Layer 2 + row-max pool: 4 n-tiles x 2 k-tiles.
#pragma unroll
  for (int nt = 0; nt < 4; ++nt) {
    v8f acc = {};
#pragma unroll
    for (int kt = 0; kt < 2; ++kt) {
      v16bf a = load_afrag(sh[wave], 64, row, khalf, kt * 32);
      v16bf b = *(const v16bf*)(fragM2 + (((size_t)kt * 4 + nt) << 9) + lane * 16);
      acc = __builtin_amdgcn_wmma_f32_16x16x32_bf16(false, a, false, b,
                                                    (short)0, acc, false, false);
    }
    float mx = 0.0f;                                  // relu lower bound
#pragma unroll
    for (int v = 0; v < 8; ++v) mx = fmaxf(mx, acc[v]);
    mx = fmaxf(mx, __shfl_xor(mx, 16, 32));           // combine row halves
    if (em) mx = 0.0f;
    if (lane < 16) pooled[(size_t)m * COUT + nt * 16 + lane] = f2bf(mx);
  }
}

// ---------------------------------------------------------------------------
// Generic bf16 WMMA GEMM: Out[128 x Ndim] = relu?(A[128 x Kdim] @ B).
// One wave per 16x16 output tile; blockDim=256 (8 waves), gridDim = ntiles.
// ---------------------------------------------------------------------------
__global__ void k_gemm(const unsigned short* __restrict__ A,
                       const unsigned short* __restrict__ Bfrag,
                       unsigned short* __restrict__ Out,
                       int Kdim, int Ndim, int do_relu) {
  int ntiles = Ndim >> 4;
  int wave = blockIdx.x * 8 + (threadIdx.x >> 5);
  int lane = threadIdx.x & 31;
  int mt = wave / ntiles, nt = wave % ntiles;
  int ktiles = Kdim >> 5;
  int row = lane & 15, khalf = lane >> 4;
  const unsigned short* Arow = A + (size_t)(mt * 16 + row) * Kdim;

  v8f acc = {};
  for (int kt = 0; kt < ktiles; ++kt) {
    if (kt + 1 < ktiles)   // -> global_prefetch_b8 for the B stream
      __builtin_prefetch((const void*)(Bfrag + (((size_t)(kt + 1) * ntiles + nt) << 9)
                                       + lane * 16), 0, 1);
    v16bf a = load_afrag(Arow, 0 /*row folded into Arow*/, 0, khalf, kt * 32);
    v16bf b = *(const v16bf*)(Bfrag + (((size_t)kt * ntiles + nt) << 9) + lane * 16);
    acc = __builtin_amdgcn_wmma_f32_16x16x32_bf16(false, a, false, b,
                                                  (short)0, acc, false, false);
  }
  int n = nt * 16 + row;
#pragma unroll
  for (int v = 0; v < 8; ++v) {
    float x = acc[v];
    if (do_relu) x = fmaxf(x, 0.0f);
    Out[(size_t)(mt * 16 + v + 8 * khalf) * Ndim + n] = f2bf(x);
  }
}

// ---------------------------------------------------------------------------
// Final heads: out[roi][0]=cls, [1]=iou, [2..8]=reg. One wave per output.
// ---------------------------------------------------------------------------
__global__ void k_heads(const unsigned short* __restrict__ hc2,
                        const unsigned short* __restrict__ hi2,
                        const unsigned short* __restrict__ hr2,
                        const float* __restrict__ wc3, const float* __restrict__ bc3,
                        const float* __restrict__ wi3, const float* __restrict__ bi3,
                        const float* __restrict__ wr3, const float* __restrict__ br3,
                        float* __restrict__ out) {
  int o = blockIdx.x * 8 + (threadIdx.x >> 5);
  int lane = threadIdx.x & 31;
  if (o >= NROI * 9) return;
  int roi = o / 9, c = o % 9;
  const unsigned short* h; const float* w3; float b; int stride;
  if (c == 0)      { h = hc2; w3 = wc3;            b = bc3[0];     stride = 1; }
  else if (c == 1) { h = hi2; w3 = wi3;            b = bi3[0];     stride = 1; }
  else             { h = hr2; w3 = wr3 + (c - 2);  b = br3[c - 2]; stride = 7; }
  float s = 0.0f;
  for (int k = lane; k < 512; k += 32)
    s += bf2f(h[(size_t)roi * 512 + k]) * w3[(size_t)k * stride];
#pragma unroll
  for (int d = 16; d; d >>= 1) s += __shfl_xor(s, d, 32);
  if (lane == 0) out[roi * 9 + c] = s + b;
}

// ---------------------------------------------------------------------------
extern "C" void kernel_launch(void* const* d_in, const int* in_sizes, int n_in,
                              void* d_out, int out_size, void* d_ws, size_t ws_size,
                              hipStream_t stream) {
  const float* xyz  = (const float*)d_in[0];
  const float* feats= (const float*)d_in[1];
  const float* rois = (const float*)d_in[2];
  const float* w_m1 = (const float*)d_in[3];
  const float* w_m2 = (const float*)d_in[4];
  const float* w_s1 = (const float*)d_in[5];
  const float* w_s2 = (const float*)d_in[6];
  const float* w_c1 = (const float*)d_in[7];
  const float* w_c2 = (const float*)d_in[8];
  const float* w_c3 = (const float*)d_in[9];
  const float* b_c3 = (const float*)d_in[10];
  const float* w_i1 = (const float*)d_in[11];
  const float* w_i2 = (const float*)d_in[12];
  const float* w_i3 = (const float*)d_in[13];
  const float* b_i3 = (const float*)d_in[14];
  const float* w_r1 = (const float*)d_in[15];
  const float* w_r2 = (const float*)d_in[16];
  const float* w_r3 = (const float*)d_in[17];
  const float* b_r3 = (const float*)d_in[18];

  char* ws = (char*)d_ws;
  size_t off = 0;
  auto alloc = [&](size_t bytes) -> void* {
    void* p = ws + off; off = (off + bytes + 255) & ~(size_t)255; return p;
  };
  float*          gp     = (float*)alloc((size_t)MTOT * 3 * 4);
  int*            idxA   = (int*)alloc((size_t)MTOT * KNN * 4);
  int*            emp    = (int*)alloc((size_t)MTOT * 4);
  unsigned short* pfeat  = (unsigned short*)alloc((size_t)NPTS * CFEAT * 2);
  unsigned short* pooled = (unsigned short*)alloc((size_t)MTOT * COUT * 2);   // == [128][13824]
  unsigned short* hid1   = (unsigned short*)alloc((size_t)NROI * 1024 * 2);
  unsigned short* sfc    = (unsigned short*)alloc((size_t)NROI * 512 * 2);
  unsigned short* hC1    = (unsigned short*)alloc((size_t)NROI * 1024 * 2);
  unsigned short* hC2    = (unsigned short*)alloc((size_t)NROI * 512 * 2);
  unsigned short* hI1    = (unsigned short*)alloc((size_t)NROI * 1024 * 2);
  unsigned short* hI2    = (unsigned short*)alloc((size_t)NROI * 512 * 2);
  unsigned short* hR1    = (unsigned short*)alloc((size_t)NROI * 1024 * 2);
  unsigned short* hR2    = (unsigned short*)alloc((size_t)NROI * 512 * 2);
  unsigned short* fM1    = (unsigned short*)alloc((size_t)5 * 4 * 512 * 2);
  unsigned short* fM2    = (unsigned short*)alloc((size_t)2 * 4 * 512 * 2);
  unsigned short* fS1    = (unsigned short*)alloc((size_t)432 * 64 * 512 * 2);  // 28.3 MB
  unsigned short* fS2    = (unsigned short*)alloc((size_t)32 * 32 * 512 * 2);
  unsigned short* fC1    = (unsigned short*)alloc((size_t)16 * 64 * 512 * 2);
  unsigned short* fC2    = (unsigned short*)alloc((size_t)32 * 32 * 512 * 2);
  unsigned short* fI1    = (unsigned short*)alloc((size_t)16 * 64 * 512 * 2);
  unsigned short* fI2    = (unsigned short*)alloc((size_t)32 * 32 * 512 * 2);
  unsigned short* fR1    = (unsigned short*)alloc((size_t)16 * 64 * 512 * 2);
  unsigned short* fR2    = (unsigned short*)alloc((size_t)32 * 32 * 512 * 2);

  // Precision / packing prep.
  hipLaunchKernelGGL(k_feats2bf, dim3((NPTS * CFEAT + 255) / 256), dim3(256), 0, stream, feats, pfeat);
  hipLaunchKernelGGL(k_frag_m1, dim3(5 * 4),    dim3(256), 0, stream, w_m1, fM1);
  hipLaunchKernelGGL(k_frag,    dim3(2 * 4),    dim3(256), 0, stream, w_m2, fM2, 64,    64,   4);
  hipLaunchKernelGGL(k_frag,    dim3(432 * 64), dim3(256), 0, stream, w_s1, fS1, 13824, 1024, 64);
  hipLaunchKernelGGL(k_frag,    dim3(32 * 32),  dim3(256), 0, stream, w_s2, fS2, 1024,  512,  32);
  hipLaunchKernelGGL(k_frag,    dim3(16 * 64),  dim3(256), 0, stream, w_c1, fC1, 512,   1024, 64);
  hipLaunchKernelGGL(k_frag,    dim3(32 * 32),  dim3(256), 0, stream, w_c2, fC2, 1024,  512,  32);
  hipLaunchKernelGGL(k_frag,    dim3(16 * 64),  dim3(256), 0, stream, w_i1, fI1, 512,   1024, 64);
  hipLaunchKernelGGL(k_frag,    dim3(32 * 32),  dim3(256), 0, stream, w_i2, fI2, 1024,  512,  32);
  hipLaunchKernelGGL(k_frag,    dim3(16 * 64),  dim3(256), 0, stream, w_r1, fR1, 512,   1024, 64);
  hipLaunchKernelGGL(k_frag,    dim3(32 * 32),  dim3(256), 0, stream, w_r2, fR2, 1024,  512,  32);

  // Geometry + neighbor search.
  hipLaunchKernelGGL(k_gridpoints, dim3((MTOT + 255) / 256), dim3(256), 0, stream, rois, gp);
  hipLaunchKernelGGL(k_neighbors,  dim3(MTOT / 8), dim3(256), 0, stream, gp, xyz, idxA, emp);

  // Point MLP + pool (WMMA, async gather-to-LDS).
  hipLaunchKernelGGL(k_pointmlp, dim3(MTOT / 8), dim3(256), 0, stream,
                     gp, xyz, pfeat, idxA, emp, fM1, fM2, pooled);

  // Shared FC + heads (WMMA GEMMs). gridDim = Ndim/16 blocks of 8 waves.
  hipLaunchKernelGGL(k_gemm, dim3(64), dim3(256), 0, stream, pooled, fS1, hid1, 13824, 1024, 1);
  hipLaunchKernelGGL(k_gemm, dim3(32), dim3(256), 0, stream, hid1,   fS2, sfc,  1024,  512,  1);
  hipLaunchKernelGGL(k_gemm, dim3(64), dim3(256), 0, stream, sfc,    fC1, hC1,  512,   1024, 1);
  hipLaunchKernelGGL(k_gemm, dim3(32), dim3(256), 0, stream, hC1,    fC2, hC2,  1024,  512,  1);
  hipLaunchKernelGGL(k_gemm, dim3(64), dim3(256), 0, stream, sfc,    fI1, hI1,  512,   1024, 1);
  hipLaunchKernelGGL(k_gemm, dim3(32), dim3(256), 0, stream, hI1,    fI2, hI2,  1024,  512,  1);
  hipLaunchKernelGGL(k_gemm, dim3(64), dim3(256), 0, stream, sfc,    fR1, hR1,  512,   1024, 1);
  hipLaunchKernelGGL(k_gemm, dim3(32), dim3(256), 0, stream, hR1,    fR2, hR2,  1024,  512,  1);

  // Final 9-way matvec + bias -> d_out (float32, [128][9]).
  hipLaunchKernelGGL(k_heads, dim3((NROI * 9 + 7) / 8), dim3(256), 0, stream,
                     hC2, hI2, hR2, w_c3, b_c3, w_i3, b_i3, w_r3, b_r3, (float*)d_out);
}
